// SelfAttention2d_9234179686400
// MI455X (gfx1250) — matrix operand order repaired and checked by reading, not compiled, not run on hardware
//
#include <hip/hip_runtime.h>
#include <string.h>

// ---------------------------------------------------------------------------
// CDNA5 / gfx1250 self-attention-2d.  wave32, WMMA bf16 (f32 accumulate),
// flash-attention (64-j chunks), double-buffered TDM async tensor loads,
// native bf16 converts, packed b32 LDS transposes, imm-offset gathers.
// ---------------------------------------------------------------------------

typedef __attribute__((ext_vector_type(16))) __bf16       v16bf;
typedef __attribute__((ext_vector_type(2)))  __bf16       v2bf;
typedef __attribute__((ext_vector_type(8)))  float        v8f;
typedef __attribute__((ext_vector_type(4)))  unsigned int su4;
typedef __attribute__((ext_vector_type(8)))  unsigned int su8;

#define DEV static __device__ __forceinline__

constexpr int B_  = 4;
constexpr int C_  = 256;
constexpr int C8_ = 32;
constexpr int C2_ = 128;
constexpr int N_  = 4096;   // H*W

union Frag16 { v16bf v; unsigned short s[16]; uint4 q[2]; };
union Acc8   { v8f   v; float f[8]; };
union Pack8  { unsigned short s[8]; uint4 q; };

// native bf16 converts (v_cvt_bf16_f32 / v_cvt_pk_bf16_f32), RTNE
DEV unsigned short f2bf(float x) {
  __bf16 b = (__bf16)x;
  return __builtin_bit_cast(unsigned short, b);
}
DEV unsigned int pack2bf(float lo, float hi) {
  v2bf p; p.x = (__bf16)lo; p.y = (__bf16)hi;
  return __builtin_bit_cast(unsigned int, p);
}

// B-fragment (16-bit, 32x16): lane half h holds k = 16*h + e -> 16 contiguous
DEV void ldB(Frag16& d, const unsigned short* p) {
  d.q[0] = *(const uint4*)(p);
  d.q[1] = *(const uint4*)(p + 8);
}
// A-fragment (16-bit, 16x32): lane half h holds k = 16*h + e -> 16 contiguous
DEV void ldA(Frag16& d, const unsigned short* row, int half) {
  d.q[0] = *(const uint4*)(row + 16 * half);
  d.q[1] = *(const uint4*)(row + 16 * half + 8);
}

DEV v8f wmma_bf16(const Frag16& a, const Frag16& b, v8f c) {
  return __builtin_amdgcn_wmma_f32_16x16x32_bf16(
      false, a.v, false, b.v, (short)0, c, false, false);
}

// ---------------------------------------------------------------------------
// Tensor Data Mover: async 2D tile load (bf16) global -> LDS with row padding.
// Descriptor per CDNA5 ISA; issued once per wave.
//   LDS rows padded: pad_interval code 4 (32 dwords = 128B) + pad_amount
//   code 3 (4 dwords = 16B) -> 144B LDS rows == 72 ushorts.
// ---------------------------------------------------------------------------
DEV void tdm_load_bf16_2d(unsigned int lds_byte_addr,
                          unsigned long long global_byte_addr,
                          unsigned int tile_d0, unsigned int tile_d1,
                          unsigned long long stride0_elems) {
  su4 g0;
  g0.x = 1u;                                     // count=1 valid user D#
  g0.y = lds_byte_addr;                          // [63:32] lds_addr
  g0.z = (unsigned int)global_byte_addr;         // [95:64] global_addr lo
  g0.w = (unsigned int)(global_byte_addr >> 32) | (2u << 30);  // type=2
  su8 g1;
  g1[0] = (1u << 16)        // data_size = 1 (2 bytes)
        | (1u << 20)        // pad_enable
        | (4u << 22)        // pad_interval: 32 dwords per row
        | (3u << 25);       // pad_amount: 4 dwords of pad
  g1[1] = (tile_d0 & 0xffffu) << 16;             // tensor_dim0[15:0]
  g1[2] = (tile_d0 >> 16) | ((tile_d1 & 0xffffu) << 16);
  g1[3] = (tile_d1 >> 16) | (tile_d0 << 16);     // | tile_dim0
  g1[4] = tile_d1;                               // tile_dim1 (tile_dim2=0)
  g1[5] = (unsigned int)stride0_elems;           // tensor_dim0_stride lo
  g1[6] = (unsigned int)(stride0_elems >> 32) & 0xffffu;
  g1[7] = 0u;
  asm volatile("tensor_load_to_lds %0, %1" : : "s"(g0), "s"(g1) : "memory");
}

// ===========================================================================
// Kernel 1: fgh projection.  fgh[o][n] = sum_c Wcat[o][c] * x[b][c][n] + bias
// grid: (N/64, 192/16, B), block 128 (4 waves)
// ===========================================================================
__global__ __launch_bounds__(128) void k_project(
    const float* __restrict__ x,
    const float* __restrict__ wf, const float* __restrict__ bfp,
    const float* __restrict__ wg, const float* __restrict__ bgp,
    const float* __restrict__ wh, const float* __restrict__ bhp,
    unsigned short* __restrict__ f_ws,
    unsigned short* __restrict__ g_ws,
    unsigned short* __restrict__ h_ws)
{
  constexpr int WSTR = 20, XSTR = 20;                 // u32 row strides (pad)
  __shared__ alignas(16) unsigned int wt32[16 * WSTR];  // [m=o][k-pair]
  __shared__ alignas(16) unsigned int xt32[64 * XSTR];  // [n][c-pair]

  const int t = threadIdx.x, lane = t & 31, w = t >> 5;
  const int half = lane >> 4, ml = lane & 15;
  const int n0 = blockIdx.x * 64;
  const int o0 = blockIdx.y * 16;
  const int bb = blockIdx.z;

  const float* Wp; const float* Bp; unsigned short* Dst; int loc, drows;
  if (o0 < 32)      { Wp = wf; Bp = bfp; Dst = f_ws; loc = o0;      drows = C8_; }
  else if (o0 < 64) { Wp = wg; Bp = bgp; Dst = g_ws; loc = o0 - 32; drows = C8_; }
  else              { Wp = wh; Bp = bhp; Dst = h_ws; loc = o0 - 64; drows = C2_; }

  Acc8 acc;
#pragma unroll
  for (int i = 0; i < 8; i++) acc.f[i] = 0.0f;

  for (int k0 = 0; k0 < C_; k0 += 32) {
    { // W tile [16m x 32k] fp32 -> packed bf16 pairs along k
      const int m = t >> 3, kq = (t & 7) * 4;
      float4 q = *(const float4*)(Wp + (size_t)(loc + m) * C_ + k0 + kq);
      wt32[m * WSTR + (kq >> 1) + 0] = pack2bf(q.x, q.y);
      wt32[m * WSTR + (kq >> 1) + 1] = pack2bf(q.z, q.w);
    }
    { // x tile [32c x 64n] -> transposed [n][c], c-pairs packed in b32
      const int cp = t & 15, ng = t >> 4;             // c = 2cp,2cp+1; 8 n's
      const float* r0 = x + ((size_t)bb * C_ + k0 + 2 * cp) * N_ + n0 + ng * 8;
      const float* r1 = r0 + N_;
      float4 a0 = ((const float4*)r0)[0], a1 = ((const float4*)r0)[1];
      float4 b0 = ((const float4*)r1)[0], b1 = ((const float4*)r1)[1];
      float va[8] = {a0.x, a0.y, a0.z, a0.w, a1.x, a1.y, a1.z, a1.w};
      float vb[8] = {b0.x, b0.y, b0.z, b0.w, b1.x, b1.y, b1.z, b1.w};
#pragma unroll
      for (int ii = 0; ii < 8; ii++)
        xt32[(ng * 8 + ii) * XSTR + cp] = pack2bf(va[ii], vb[ii]);
    }
    __syncthreads();

    Frag16 A, Bfr;
    ldA(A, (const unsigned short*)&wt32[ml * WSTR], half);
    ldB(Bfr, (const unsigned short*)&xt32[(w * 16 + ml) * XSTR] + 16 * half);
    acc.v = wmma_bf16(A, Bfr, acc.v);
    __syncthreads();
  }

  // epilogue: one base pointer, constant v*N_ immediate offsets
  unsigned short* db =
      Dst + ((size_t)bb * drows + loc + 8 * half) * N_ + n0 + w * 16 + ml;
  const float* bp = Bp + loc + 8 * half;
#pragma unroll
  for (int v = 0; v < 8; v++) db[v * N_] = f2bf(acc.f[v] + bp[v]);
}

// ===========================================================================
// Kernel 2: flash attention, 64-j chunks, double-buffered TDM h tiles.
// grid: (N/64, B), block 128 (4 waves); wave w owns a 16-row i-tile and
// issues the TDM for its own 32-row h slab one chunk ahead.
// ===========================================================================
__global__ __launch_bounds__(128) void k_attn(
    const unsigned short* __restrict__ f_ws,
    const unsigned short* __restrict__ g_ws,
    const unsigned short* __restrict__ h_ws,
    unsigned short* __restrict__ o_ws)
{
  constexpr int GSTR = 20;    // u32 stride: 40 ushorts per j-row
  constexpr int HSTR = 72;    // ushorts: 64 data + 8 pad (matches TDM pad)
  constexpr int PSTR = 72;
  constexpr int HBUF = C2_ * HSTR;                        // one h buffer
  __shared__ alignas(16)  unsigned int   gt32[64 * GSTR];     // [j][c-pair]
  __shared__ alignas(128) unsigned short ht[2 * HBUF];        // double buffer
  __shared__ alignas(16)  unsigned short pst[4 * 16 * PSTR];  // per-wave P

  const int t = threadIdx.x, lane = t & 31, w = t >> 5;
  const int half = lane >> 4, ml = lane & 15;
  const int bb = blockIdx.y;
  const int i0 = blockIdx.x * 64 + w * 16;

  // persistent f A-fragment: base + constant element offsets (imm24)
  Frag16 fA;
  {
    const unsigned short* fb =
        f_ws + ((size_t)bb * C8_ + 16 * half) * N_ + i0 + ml;
#pragma unroll
    for (int e = 0; e < 16; e++)
      fA.s[e] = fb[e * N_];   // k = 16h+e
  }

  Acc8 oacc[8];
#pragma unroll
  for (int ct = 0; ct < 8; ct++)
#pragma unroll
    for (int v = 0; v < 8; v++) oacc[ct].f[v] = 0.0f;

  float m_row[8], l_row[8];
#pragma unroll
  for (int v = 0; v < 8; v++) { m_row[v] = -3.0e38f; l_row[v] = 0.0f; }

  Acc8 zero;
#pragma unroll
  for (int v = 0; v < 8; v++) zero.f[v] = 0.0f;

  unsigned short* prow_base = &pst[w * 16 * PSTR];
  const unsigned int ht_lds = (unsigned int)(unsigned long long)(&ht[0]);
  const unsigned long long hslab =
      (unsigned long long)(const void*)(h_ws + ((size_t)bb * C2_ + 32 * w) * N_);

  constexpr int NCH = N_ / 64;
  // prologue: DMA chunk 0 into buffer 0
  tdm_load_bf16_2d(ht_lds + (unsigned int)(w * 32 * HSTR * 2),
                   hslab, 64, 32, N_);

  for (int ch = 0; ch < NCH; ch++) {
    const int p = ch & 1;
    const int j0 = ch * 64;

    // pipeline: DMA chunk ch+1 into the other buffer (reads of it finished
    // at the end-of-iteration barrier of ch-1)
    if (ch + 1 < NCH)
      tdm_load_bf16_2d(
          ht_lds + (unsigned int)((p ^ 1) * HBUF * 2 + w * 32 * HSTR * 2),
          hslab + (size_t)(j0 + 64) * 2, 64, 32, N_);

    // stage g tile [32c x 64j] -> transposed [j][c], c-pairs packed b32
    {
      const int cp = t & 15, jg = t >> 4;  // c = 2cp,2cp+1 ; j = jg*8..+7
      const unsigned short* r0 =
          g_ws + ((size_t)bb * C8_ + 2 * cp) * N_ + j0 + jg * 8;
      uint4 d0 = *(const uint4*)r0;
      uint4 d1 = *(const uint4*)(r0 + N_);
      const unsigned short* s0p = (const unsigned short*)&d0;
      const unsigned short* s1p = (const unsigned short*)&d1;
#pragma unroll
      for (int jj = 0; jj < 8; jj++)
        gt32[(jg * 8 + jj) * GSTR + cp] =
            (unsigned int)s0p[jj] | ((unsigned int)s1p[jj] << 16);
      if (j0 + 64 < N_)  // global_prefetch_b8 of next g chunk
        __builtin_prefetch(r0 + 64, 0, 1);
    }

    // per-wave TDM ops are in order: <=1 outstanding means chunk ch landed
    if (ch + 1 < NCH) __builtin_amdgcn_s_wait_tensorcnt(1);
    else              __builtin_amdgcn_s_wait_tensorcnt(0);
    __syncthreads();                        // all slabs + gt visible

    // --- scores: 4 j-tiles, K = C8 = 32
    Acc8 s[4];
#pragma unroll
    for (int jt = 0; jt < 4; jt++) {
      Frag16 gB;
      ldB(gB, (const unsigned short*)&gt32[(jt * 16 + ml) * GSTR] + 16 * half);
      s[jt].v = wmma_bf16(fA, gB, zero.v);
    }

    // --- online softmax per row slot (row i = v + 8*half)
#pragma unroll
    for (int v = 0; v < 8; v++) {
      float a = fmaxf(fmaxf(s[0].f[v], s[1].f[v]),
                      fmaxf(s[2].f[v], s[3].f[v]));
#pragma unroll
      for (int off = 8; off; off >>= 1) a = fmaxf(a, __shfl_xor(a, off, 32));
      const float mnew = fmaxf(m_row[v], a);
      const float corr = __expf(m_row[v] - mnew);
      float p4[4], rs = 0.0f;
#pragma unroll
      for (int jt = 0; jt < 4; jt++) {
        p4[jt] = __expf(s[jt].f[v] - mnew); rs += p4[jt];
      }
#pragma unroll
      for (int off = 8; off; off >>= 1) rs += __shfl_xor(rs, off, 32);
      l_row[v] = l_row[v] * corr + rs;
      m_row[v] = mnew;
#pragma unroll
      for (int ct = 0; ct < 8; ct++) oacc[ct].f[v] *= corr;
      unsigned short* pr = prow_base + (v + 8 * half) * PSTR;
#pragma unroll
      for (int jt = 0; jt < 4; jt++) pr[jt * 16 + ml] = f2bf(p4[jt]);
    }
    // NOTE: no block barrier needed here -- pst is per-wave and same-wave
    // LDS ops complete in order (ISA 7.3 "LDS: in order").

    // --- PV: oacc += P[16i x 64j] * h^T[64j x 16c]  (two K=32 halves)
    const unsigned short* hb = &ht[p * HBUF];
#pragma unroll
    for (int kk = 0; kk < 2; kk++) {
      Frag16 pA;
      ldA(pA, prow_base + ml * PSTR + 32 * kk, half);
#pragma unroll
      for (int ct = 0; ct < 8; ct++) {
        Frag16 hB;
        ldB(hB, &hb[(ct * 16 + ml) * HSTR + 32 * kk + 16 * half]);
        oacc[ct].v = wmma_bf16(pA, hB, oacc[ct].v);
      }
    }
    __syncthreads();   // all reads of buffer p / gt done before reuse
  }

  // epilogue: normalize, pack 8 consecutive i, imm-offset b128 stores
  unsigned short* ob = o_ws + ((size_t)bb * C2_ + ml) * N_ + i0 + 8 * half;
#pragma unroll
  for (int ct = 0; ct < 8; ct++) {
    Pack8 pk;
#pragma unroll
    for (int v = 0; v < 8; v++) pk.s[v] = f2bf(oacc[ct].f[v] / l_row[v]);
    *(uint4*)(ob + ct * 16 * N_) = pk.q;
  }
}

// ===========================================================================
// Kernel 3: out = gamma * (wv[256x128] @ o + bv) + x     (K = 128)
// grid: (N/64, 256/16, B), block 128 (4 waves)
// ===========================================================================
__global__ __launch_bounds__(128) void k_outproj(
    const unsigned short* __restrict__ o_ws,
    const float* __restrict__ wv, const float* __restrict__ bv,
    const float* __restrict__ gamma,
    const float* __restrict__ x,
    float* __restrict__ out)
{
  constexpr int WSTR = 20, OSTR = 20;
  __shared__ alignas(16) unsigned int wt32[16 * WSTR];  // [m=ch][k-pair]
  __shared__ alignas(16) unsigned int ot32[64 * OSTR];  // [n][c-pair]

  const int t = threadIdx.x, lane = t & 31, w = t >> 5;
  const int half = lane >> 4, ml = lane & 15;
  const int n0 = blockIdx.x * 64;
  const int o0 = blockIdx.y * 16;
  const int bb = blockIdx.z;

  Acc8 acc;
#pragma unroll
  for (int i = 0; i < 8; i++) acc.f[i] = 0.0f;

  for (int k0 = 0; k0 < C2_; k0 += 32) {
    {
      const int m = t >> 3, kq = (t & 7) * 4;
      float4 q = *(const float4*)(wv + (size_t)(o0 + m) * C2_ + k0 + kq);
      wt32[m * WSTR + (kq >> 1) + 0] = pack2bf(q.x, q.y);
      wt32[m * WSTR + (kq >> 1) + 1] = pack2bf(q.z, q.w);
    }
    {
      const int cp = t & 15, ng = t >> 4;   // c = 2cp,2cp+1 ; n = ng*8..+7
      const unsigned short* r0 =
          o_ws + ((size_t)bb * C2_ + k0 + 2 * cp) * N_ + n0 + ng * 8;
      uint4 d0 = *(const uint4*)r0;
      uint4 d1 = *(const uint4*)(r0 + N_);
      const unsigned short* s0p = (const unsigned short*)&d0;
      const unsigned short* s1p = (const unsigned short*)&d1;
#pragma unroll
      for (int ii = 0; ii < 8; ii++)
        ot32[(ng * 8 + ii) * OSTR + cp] =
            (unsigned int)s0p[ii] | ((unsigned int)s1p[ii] << 16);
    }
    __syncthreads();

    Frag16 A, Bfr;
    ldA(A, (const unsigned short*)&wt32[ml * WSTR], half);
    ldB(Bfr, (const unsigned short*)&ot32[(w * 16 + ml) * OSTR] + 16 * half);
    acc.v = wmma_bf16(A, Bfr, acc.v);
    __syncthreads();
  }

  const float gm = gamma[0];
  const float* bvp = bv + o0 + 8 * half;
  const float* xb  = x   + ((size_t)bb * C_ + o0 + 8 * half) * N_ + n0 + w * 16 + ml;
  float*       ub  = out + ((size_t)bb * C_ + o0 + 8 * half) * N_ + n0 + w * 16 + ml;
#pragma unroll
  for (int v = 0; v < 8; v++)
    ub[v * N_] = gm * (acc.f[v] + bvp[v]) + xb[v * N_];
}

// ===========================================================================
extern "C" void kernel_launch(void* const* d_in, const int* in_sizes, int n_in,
                              void* d_out, int out_size, void* d_ws, size_t ws_size,
                              hipStream_t stream) {
  (void)in_sizes; (void)n_in; (void)out_size; (void)ws_size;
  const float* x     = (const float*)d_in[0];
  const float* wf    = (const float*)d_in[1];
  const float* bfp   = (const float*)d_in[2];
  const float* wg    = (const float*)d_in[3];
  const float* bgp   = (const float*)d_in[4];
  const float* wh    = (const float*)d_in[5];
  const float* bhp   = (const float*)d_in[6];
  const float* wv    = (const float*)d_in[7];
  const float* bv    = (const float*)d_in[8];
  const float* gamma = (const float*)d_in[9];
  float* out = (float*)d_out;

  unsigned short* f_ws = (unsigned short*)d_ws;
  unsigned short* g_ws = f_ws + (size_t)B_ * C8_ * N_;
  unsigned short* h_ws = g_ws + (size_t)B_ * C8_ * N_;
  unsigned short* o_ws = h_ws + (size_t)B_ * C2_ * N_;

  k_project<<<dim3(N_ / 64, 192 / 16, B_), 128, 0, stream>>>(
      x, wf, bfp, wg, bgp, wh, bhp, f_ws, g_ws, h_ws);
  k_attn<<<dim3(N_ / 64, B_), 128, 0, stream>>>(f_ws, g_ws, h_ws, o_ws);
  k_outproj<<<dim3(N_ / 64, C_ / 16, B_), 128, 0, stream>>>(
      o_ws, wv, bv, gamma, x, out);
}
